// TransformerEncoderLayer_5428838662584
// MI455X (gfx1250) — compile-verified, matrix operand order checked by reading, not run on hardware
//
#include <hip/hip_runtime.h>
#include <hip/hip_bf16.h>
#include <cstddef>
#include <cstdint>

// ---------------------------------------------------------------------------
// Transformer-XL encoder layer for MI455X (gfx1250, wave32).
// bf16 WMMA everywhere; TDM (tensor_load_to_lds) stages GEMM operand panels
// into LDS with double buffering; all tensors converted to bf16 once at
// production so no f32->bf16 cvt remains in GEMM inner loops.
// ---------------------------------------------------------------------------

typedef __bf16 bf16_t;
typedef bf16_t v16bf __attribute__((ext_vector_type(16)));
typedef float  v8f   __attribute__((ext_vector_type(8)));
typedef unsigned int u32x4 __attribute__((ext_vector_type(4)));
typedef int          i32x8 __attribute__((ext_vector_type(8)));
typedef int          i32x4 __attribute__((ext_vector_type(4)));

__device__ __forceinline__ v8f vzero8() {
    v8f x;
    #pragma unroll
    for (int i = 0; i < 8; ++i) x[i] = 0.0f;
    return x;
}

__device__ __forceinline__ v8f wmma_bf16(v16bf a, v16bf b, v8f c) {
    return __builtin_amdgcn_wmma_f32_16x16x32_bf16(
        false, a, false, b, (short)0, c, false, false);
}

// A/B fragment (16x32 bf16) per ISA layout: lane L -> row = L%16,
// k-half = L/16; pair 2j,2j+1 at k = half*8 + 2j (+16 for j>=4).
__device__ __forceinline__ v16bf frag_rm_bf16(const bf16_t* base, int ld,
                                              int row0, int k0, int lane) {
    const bf16_t* p = base + (size_t)(row0 + (lane & 15)) * ld
                           + k0 + ((lane >> 4) << 3);
    v16bf f;
    #pragma unroll
    for (int j = 0; j < 4; ++j) {
        f[2 * j]     = p[2 * j];
        f[2 * j + 1] = p[2 * j + 1];
        f[8 + 2 * j] = p[16 + 2 * j];
        f[9 + 2 * j] = p[17 + 2 * j];
    }
    return f;
}

// Same fragment from f32 source (LDS score strip), converting.
__device__ __forceinline__ v16bf frag_rm_f32(const float* base, int ld,
                                             int row0, int k0, int lane) {
    const float* p = base + (size_t)(row0 + (lane & 15)) * ld
                          + k0 + ((lane >> 4) << 3);
    v16bf f;
    #pragma unroll
    for (int j = 0; j < 4; ++j) {
        float2 x0 = *(const float2*)(p + 2 * j);
        float2 x1 = *(const float2*)(p + 16 + 2 * j);
        f[2 * j]     = (bf16_t)x0.x;
        f[2 * j + 1] = (bf16_t)x0.y;
        f[8 + 2 * j] = (bf16_t)x1.x;
        f[9 + 2 * j] = (bf16_t)x1.y;
    }
    return f;
}

// B fragment where B[k][n] = base[k*ld + n] (k-major, e.g. V tensor).
__device__ __forceinline__ v16bf frag_cs_bf16(const bf16_t* base, int ld,
                                              int k0, int col0, int lane) {
    const bf16_t* p = base + (size_t)(k0 + ((lane >> 4) << 3)) * ld
                           + col0 + (lane & 15);
    v16bf f;
    #pragma unroll
    for (int j = 0; j < 4; ++j) {
        f[2 * j]     = p[(size_t)(2 * j) * ld];
        f[2 * j + 1] = p[(size_t)(2 * j + 1) * ld];
        f[8 + 2 * j] = p[(size_t)(16 + 2 * j) * ld];
        f[9 + 2 * j] = p[(size_t)(17 + 2 * j) * ld];
    }
    return f;
}

// ---------------------------------------------------------------------------
// TDM: 2D tile (tile_k bf16 elems contiguous, tile_rows rows, row stride
// strideElems) from global -> LDS.  D# packed per CDNA5 ISA ch.8.
// This toolchain's builtin is the 6-arg form:
//   (u32x4 g0, i32x8 g1, i32x4 g2, i32x4 g3, i32x8 g4, i32 cpol)
// ---------------------------------------------------------------------------
__device__ __forceinline__ void tdm_load_2d(bf16_t* lds_ptr, const bf16_t* g,
                                            unsigned tk, unsigned tr,
                                            unsigned strideElems) {
    const unsigned lds_off = (unsigned)(uintptr_t)lds_ptr;  // addr[31:0]=LDS off
    const unsigned long long ga = (unsigned long long)(uintptr_t)g;
    u32x4 g0;
    g0[0] = 1u;                                   // count=1, user mode
    g0[1] = lds_off;                              // lds_addr (bytes)
    g0[2] = (unsigned)ga;                         // global_addr[31:0]
    g0[3] = (unsigned)((ga >> 32) & 0x1ffffffull) // global_addr[56:32]
          | 0x80000000u;                          // type=2 ("image")
    i32x8 g1;
    g1[0] = 0x00010000;          // data_size=1 (2 bytes); no multicast/pad
    g1[1] = (int)(tk << 16);     // tensor_dim0[15:0]   (= tile: in-bounds)
    g1[2] = (int)(tr << 16);     // tensor_dim1[15:0]
    g1[3] = (int)(tk << 16);     // tile_dim0
    g1[4] = (int)tr;             // tile_dim1 (tile_dim2 = 0)
    g1[5] = (int)strideElems;    // tensor_dim0_stride[31:0]
    g1[6] = 0;
    g1[7] = 0;
    i32x4 g2;
    g2[0] = 0; g2[1] = 0; g2[2] = 0; g2[3] = 0;
    i32x4 g3 = g2;
    i32x8 g4;
    #pragma unroll
    for (int i = 0; i < 8; ++i) g4[i] = 0;
    __builtin_amdgcn_tensor_load_to_lds(g0, g1, g2, g3, g4, 0);
}

// ---------------------------------------------------------------------------
// WG-cooperative GEMM: C[M,N] = epi(A[M,K] @ W[N,K]^T + bias (+R)).
// Block = 256 thr (8 waves), WG tile 128x64.  Wave w owns rows w*16..w*16+15.
// TDM double-buffers A(128x64) and B(64x64) bf16 panels in LDS.
// EPI: 0=bias, 1=bias+ReLU, 2=bias+residual.  OUTF/OUTB: f32/bf16 outputs.
// ---------------------------------------------------------------------------
template <int EPI, bool OUTF, bool OUTB>
__global__ __launch_bounds__(256)
void gemm_tdm(const bf16_t* __restrict__ A, const bf16_t* __restrict__ W,
              const float* __restrict__ bias, const float* __restrict__ R,
              float* __restrict__ Cf, bf16_t* __restrict__ Cb,
              int M, int N, int K) {
    __shared__ bf16_t lA[2][128 * 64];
    __shared__ bf16_t lB[2][64 * 64];
    const int lane = threadIdx.x & 31;
    const int wave = threadIdx.x >> 5;
    const int n0 = blockIdx.x * 64;
    const int m0 = blockIdx.y * 128;

    v8f acc[4];
    #pragma unroll
    for (int j = 0; j < 4; ++j) acc[j] = vzero8();

    const int nk = K >> 6;
    if (wave == 0) {   // wave-uniform guard; TDM ignores EXEC, one issue per WG
        tdm_load_2d(&lA[0][0], A + (size_t)m0 * K, 64, 128, (unsigned)K);
        tdm_load_2d(&lB[0][0], W + (size_t)n0 * K, 64, 64, (unsigned)K);
    }
    for (int kk = 0; kk < nk; ++kk) {
        const int buf = kk & 1;
        if (wave == 0) {
            if (kk + 1 < nk) {   // prefetch next slab, then wait current (<=2)
                const int k1 = (kk + 1) << 6;
                tdm_load_2d(&lA[buf ^ 1][0], A + (size_t)m0 * K + k1, 64, 128,
                            (unsigned)K);
                tdm_load_2d(&lB[buf ^ 1][0], W + (size_t)n0 * K + k1, 64, 64,
                            (unsigned)K);
                __builtin_amdgcn_s_wait_tensorcnt(2);
            } else {
                __builtin_amdgcn_s_wait_tensorcnt(0);
            }
        }
        __syncthreads();
        #pragma unroll
        for (int ks = 0; ks < 64; ks += 32) {
            v16bf a = frag_rm_bf16(&lA[buf][0], 64, wave * 16, ks, lane);
            #pragma unroll
            for (int j = 0; j < 4; ++j) {
                v16bf b = frag_rm_bf16(&lB[buf][0], 64, j * 16, ks, lane);
                acc[j] = wmma_bf16(a, b, acc[j]);
            }
        }
        __syncthreads();   // panels free before next TDM overwrites them
    }

    const int nl = lane & 15, half = lane >> 4;
    #pragma unroll
    for (int j = 0; j < 4; ++j) {
        const int n = n0 + j * 16 + nl;
        const float bv = bias[n];
        #pragma unroll
        for (int r = 0; r < 8; ++r) {
            const int m = m0 + wave * 16 + r + half * 8;
            float vc = acc[j][r] + bv;
            if (EPI == 1) vc = fmaxf(vc, 0.0f);
            if (EPI == 2) vc += R[(size_t)m * N + n];
            if (OUTF) Cf[(size_t)m * N + n] = vc;
            if (OUTB) Cb[(size_t)m * N + n] = (bf16_t)vc;
        }
    }
}

// ---------------------------------------------------------------------------
// Fused relative attention (bf16 operands).  One WG = (b, h, 32-query block).
// sc[32][1024] f32 score strip in LDS; _rel_shift applied as exact scatter:
//   unshifted x[q2][c] -> (row q2, col c-(S-1-q2))  if c >= S-1-q2
//                      -> (row q2-1, col c+q2+1)    otherwise
// ---------------------------------------------------------------------------
__global__ __launch_bounds__(256)
void attn_fused(const bf16_t* __restrict__ Q, const bf16_t* __restrict__ Kb,
                const bf16_t* __restrict__ Vb, const bf16_t* __restrict__ Rel,
                const float* __restrict__ rwb, const float* __restrict__ rrb,
                bf16_t* __restrict__ O) {
    const int S = 1024, Dm = 1024, Dh = 64, QB = 32;
    extern __shared__ float smem[];
    float*  sc    = smem;                     // [32][1024]
    float*  extra = sc + QB * S;              // [1024]
    bf16_t* qrw   = (bf16_t*)(extra + S);     // [32][64]
    bf16_t* qrr   = qrw + QB * Dh;            // [32][64]

    const int bid  = blockIdx.x;
    const int qblk = bid & 31;
    const int h    = (bid >> 5) & 15;
    const int b    = bid >> 9;
    const int q0   = qblk * QB;
    const int tid  = threadIdx.x, lane = tid & 31, wave = tid >> 5;
    const float scale = 1.0f / 32.0f;         // 1/sqrt(d_model)

    for (int idx = tid; idx < QB * Dh; idx += 256) {
        const int r = idx >> 6, d = idx & 63;
        const float qv = (float)Q[(size_t)(b * S + q0 + r) * Dm + h * Dh + d];
        qrw[idx] = (bf16_t)(qv + rwb[h * Dh + d]);
        qrr[idx] = (bf16_t)(qv + rrb[h * Dh + d]);
    }
    const int qe = q0 + QB;                   // borrowed query row
    for (int j = tid; j < S; j += 256) {
        float a = 0.0f;
        if (qe < S) {
            #pragma unroll 8
            for (int d = 0; d < Dh; ++d) {
                const float qv = (float)Q[(size_t)(b * S + qe) * Dm + h * Dh + d]
                               + rrb[h * Dh + d];
                a += qv * (float)Rel[(size_t)j * Dm + h * Dh + d];
            }
        }
        extra[j] = a;
    }
    __syncthreads();

    // scores1 = (q+rwb) @ K^T
    for (int t = 0; t < 8; ++t) {
        const int n0 = (wave * 8 + t) * 16;
        v8f acc0 = vzero8(), acc1 = vzero8();
        #pragma unroll
        for (int k0 = 0; k0 < Dh; k0 += 32) {
            v16bf a0 = frag_rm_bf16(qrw, Dh, 0, k0, lane);
            v16bf a1 = frag_rm_bf16(qrw, Dh, 16, k0, lane);
            v16bf bf = frag_rm_bf16(Kb + (size_t)(b * S + n0) * Dm + h * Dh,
                                    Dm, 0, k0, lane);
            acc0 = wmma_bf16(a0, bf, acc0);
            acc1 = wmma_bf16(a1, bf, acc1);
        }
        const int nl = lane & 15, half = lane >> 4;
        #pragma unroll
        for (int r = 0; r < 8; ++r) {
            sc[(r + half * 8) * S + n0 + nl]      = acc0[r];
            sc[(16 + r + half * 8) * S + n0 + nl] = acc1[r];
        }
    }
    __syncthreads();

    // scores2 with rel_shift scatter-add
    for (int t = 0; t < 8; ++t) {
        const int j0 = (wave * 8 + t) * 16;
        v8f acc0 = vzero8(), acc1 = vzero8();
        #pragma unroll
        for (int k0 = 0; k0 < Dh; k0 += 32) {
            v16bf a0 = frag_rm_bf16(qrr, Dh, 0, k0, lane);
            v16bf a1 = frag_rm_bf16(qrr, Dh, 16, k0, lane);
            v16bf bf = frag_rm_bf16(Rel + h * Dh, Dm, j0, k0, lane);
            acc0 = wmma_bf16(a0, bf, acc0);
            acc1 = wmma_bf16(a1, bf, acc1);
        }
        const int nl = lane & 15, half = lane >> 4;
        #pragma unroll
        for (int i = 0; i < 2; ++i) {
            const v8f& a = i ? acc1 : acc0;
            #pragma unroll
            for (int r = 0; r < 8; ++r) {
                const int ql = i * 16 + r + half * 8;
                const int qq = q0 + ql;
                const int c  = j0 + nl;
                const float v = a[r];
                if (c >= S - 1 - qq) {
                    sc[ql * S + (c - (S - 1 - qq))] += v;
                } else if (ql > 0) {
                    sc[(ql - 1) * S + (c + qq + 1)] += v;
                }
            }
        }
    }
    for (int c = tid; c < S; c += 256) {       // borrowed row -> local row 31
        const int kk = c + qe + 1;
        if (kk < S) sc[(QB - 1) * S + kk] += extra[c];
    }
    __syncthreads();

    // scale + softmax
    for (int rr = wave; rr < QB; rr += 8) {
        float* row = sc + rr * S;
        float mx = -1e30f;
        for (int c = lane; c < S; c += 32) mx = fmaxf(mx, row[c] * scale);
        #pragma unroll
        for (int o = 16; o > 0; o >>= 1) mx = fmaxf(mx, __shfl_xor(mx, o, 32));
        float sum = 0.0f;
        for (int c = lane; c < S; c += 32) {
            const float e = __expf(row[c] * scale - mx);
            row[c] = e;
            sum += e;
        }
        #pragma unroll
        for (int o = 16; o > 0; o >>= 1) sum += __shfl_xor(sum, o, 32);
        const float inv = 1.0f / sum;
        for (int c = lane; c < S; c += 32) row[c] *= inv;
    }
    __syncthreads();

    // out = attn @ V -> bf16
    {
        const int mt = wave >> 2, nt = wave & 3;
        v8f acc = vzero8();
        for (int k0 = 0; k0 < S; k0 += 32) {
            v16bf a  = frag_rm_f32(sc, S, mt * 16, k0, lane);
            v16bf bf = frag_cs_bf16(Vb + (size_t)(b * S) * Dm + h * Dh,
                                    Dm, k0, nt * 16, lane);
            acc = wmma_bf16(a, bf, acc);
        }
        const int nl = lane & 15, half = lane >> 4;
        #pragma unroll
        for (int r = 0; r < 8; ++r) {
            const int m = mt * 16 + r + half * 8;
            O[(size_t)(b * S + q0 + m) * Dm + h * Dh + nt * 16 + nl] =
                (bf16_t)acc[r];
        }
    }
}

// ---------------------------------------------------------------------------
template <bool WB>
__global__ __launch_bounds__(256)
void layernorm_row(const float* __restrict__ X, const float* __restrict__ g,
                   const float* __restrict__ bb, float* __restrict__ Y,
                   bf16_t* __restrict__ Yb, int D) {
    __shared__ float red[8];
    const int row = blockIdx.x;
    const float* x = X + (size_t)row * D;
    float s = 0.0f;
    for (int c = threadIdx.x; c < D; c += 256) s += x[c];
    #pragma unroll
    for (int o = 16; o > 0; o >>= 1) s += __shfl_xor(s, o, 32);
    if ((threadIdx.x & 31) == 0) red[threadIdx.x >> 5] = s;
    __syncthreads();
    float tot = 0.0f;
    #pragma unroll
    for (int i = 0; i < 8; ++i) tot += red[i];
    const float mu = tot / D;
    __syncthreads();
    float v2 = 0.0f;
    for (int c = threadIdx.x; c < D; c += 256) {
        const float d = x[c] - mu;
        v2 += d * d;
    }
    #pragma unroll
    for (int o = 16; o > 0; o >>= 1) v2 += __shfl_xor(v2, o, 32);
    if ((threadIdx.x & 31) == 0) red[threadIdx.x >> 5] = v2;
    __syncthreads();
    float tv = 0.0f;
    #pragma unroll
    for (int i = 0; i < 8; ++i) tv += red[i];
    const float inv = rsqrtf(tv / D + 1e-5f);
    float* y = Y + (size_t)row * D;
    for (int c = threadIdx.x; c < D; c += 256) {
        const float o = (x[c] - mu) * inv * g[c] + bb[c];
        y[c] = o;
        if (WB) Yb[(size_t)row * D + c] = (bf16_t)o;
    }
}

// ---------------------------------------------------------------------------
__global__ __launch_bounds__(256)
void cvt_f32_to_bf16(const float* __restrict__ x, bf16_t* __restrict__ y,
                     int n) {
    const int i = (blockIdx.x * 256 + threadIdx.x) * 4;
    if (i + 3 < n) {
        const float4 v = *(const float4*)(x + i);
        y[i]     = (bf16_t)v.x;
        y[i + 1] = (bf16_t)v.y;
        y[i + 2] = (bf16_t)v.z;
        y[i + 3] = (bf16_t)v.w;
    }
}

// ---------------------------------------------------------------------------
extern "C" void kernel_launch(void* const* d_in, const int* in_sizes, int n_in,
                              void* d_out, int out_size, void* d_ws,
                              size_t ws_size, hipStream_t stream) {
    const float* src  = (const float*)d_in[0];
    const float* Wq   = (const float*)d_in[1];
    const float* bq   = (const float*)d_in[2];
    const float* Wk   = (const float*)d_in[3];
    const float* bk   = (const float*)d_in[4];
    const float* Wv   = (const float*)d_in[5];
    const float* bv   = (const float*)d_in[6];
    const float* Wo   = (const float*)d_in[7];
    const float* bo   = (const float*)d_in[8];
    const float* W1   = (const float*)d_in[9];
    const float* b1   = (const float*)d_in[10];
    const float* W2   = (const float*)d_in[11];
    const float* b2   = (const float*)d_in[12];
    const float* rwb  = (const float*)d_in[13];
    const float* rrb  = (const float*)d_in[14];
    const float* ln1g = (const float*)d_in[15];
    const float* ln1b = (const float*)d_in[16];
    const float* ln2g = (const float*)d_in[17];
    const float* ln2b = (const float*)d_in[18];
    const float* pos  = (const float*)d_in[19];

    const int M = 4096, D = 1024, Dff = 4096, S = 1024;
    const size_t MD = (size_t)M * D;       // 4,194,304
    const size_t SD = (size_t)S * D;       // 1,048,576

    bf16_t* srcb  = (bf16_t*)d_ws;
    bf16_t* posb  = srcb + MD;
    bf16_t* wqb   = posb + SD;
    bf16_t* wkb   = wqb + SD;
    bf16_t* wvb   = wkb + SD;
    bf16_t* wob   = wvb + SD;
    bf16_t* w1b   = wob + SD;
    bf16_t* w2b   = w1b + MD;
    bf16_t* qb    = w2b + MD;
    bf16_t* kb    = qb + MD;
    bf16_t* vb    = kb + MD;
    bf16_t* relb  = vb + MD;
    bf16_t* attnb = relb + SD;
    bf16_t* x1b   = attnb + MD;
    bf16_t* ff1b  = x1b + MD;
    float*  pre1  = (float*)(ff1b + (size_t)M * Dff);
    float*  x1f   = pre1 + MD;
    float*  pre2  = pre1;                   // pre1 dead after LN1

    const dim3 blk(256);
    // Convert static inputs to bf16 once.
    cvt_f32_to_bf16<<<(int)(MD / 1024), blk, 0, stream>>>(src, srcb, (int)MD);
    cvt_f32_to_bf16<<<(int)(SD / 1024), blk, 0, stream>>>(pos, posb, (int)SD);
    cvt_f32_to_bf16<<<(int)(SD / 1024), blk, 0, stream>>>(Wq, wqb, (int)SD);
    cvt_f32_to_bf16<<<(int)(SD / 1024), blk, 0, stream>>>(Wk, wkb, (int)SD);
    cvt_f32_to_bf16<<<(int)(SD / 1024), blk, 0, stream>>>(Wv, wvb, (int)SD);
    cvt_f32_to_bf16<<<(int)(SD / 1024), blk, 0, stream>>>(Wo, wob, (int)SD);
    cvt_f32_to_bf16<<<(int)(MD / 1024), blk, 0, stream>>>(W1, w1b, (int)MD);
    cvt_f32_to_bf16<<<(int)(MD / 1024), blk, 0, stream>>>(W2, w2b, (int)MD);

    // Projections (bf16 out feeds attention).
    gemm_tdm<0, false, true><<<dim3(16, 32), blk, 0, stream>>>(
        srcb, wqb, bq, nullptr, nullptr, qb, M, D, D);
    gemm_tdm<0, false, true><<<dim3(16, 32), blk, 0, stream>>>(
        srcb, wkb, bk, nullptr, nullptr, kb, M, D, D);
    gemm_tdm<0, false, true><<<dim3(16, 32), blk, 0, stream>>>(
        srcb, wvb, bv, nullptr, nullptr, vb, M, D, D);
    gemm_tdm<0, false, true><<<dim3(16, 8), blk, 0, stream>>>(
        posb, wkb, bk, nullptr, nullptr, relb, S, D, D);

    // Fused attention: 4 * 16 * 32 = 2048 WGs, ~140 KB dynamic LDS.
    const size_t shb = (size_t)(32 * 1024 + 1024) * sizeof(float)
                     + (size_t)2 * 32 * 64 * sizeof(bf16_t);
    attn_fused<<<2048, blk, shb, stream>>>(qb, kb, vb, relb, rwb, rrb, attnb);

    // Wo projection + residual(src) -> LN1 (f32 + bf16 outs).
    gemm_tdm<2, true, false><<<dim3(16, 32), blk, 0, stream>>>(
        attnb, wob, bo, src, pre1, nullptr, M, D, D);
    layernorm_row<true><<<4096, blk, 0, stream>>>(pre1, ln1g, ln1b, x1f, x1b, D);

    // FFN + residual(x1) -> LN2 -> out.
    gemm_tdm<1, false, true><<<dim3(64, 32), blk, 0, stream>>>(
        x1b, w1b, b1, nullptr, nullptr, ff1b, M, Dff, D);
    gemm_tdm<2, true, false><<<dim3(16, 32), blk, 0, stream>>>(
        ff1b, w2b, b2, x1f, pre2, nullptr, M, D, Dff);
    layernorm_row<false><<<4096, blk, 0, stream>>>(pre2, ln2g, ln2b,
                                                   (float*)d_out, nullptr, D);
}
